// PeepholeLSTM_85495618994515
// MI455X (gfx1250) — matrix operand order repaired and checked by reading, not compiled, not run on hardware
//
#include <hip/hip_runtime.h>
#include <hip/hip_bf16.h>

typedef __bf16 bf16;
typedef __attribute__((ext_vector_type(16))) __bf16 v16bf;
typedef __attribute__((ext_vector_type(8)))  __bf16 v8bf;
typedef __attribute__((ext_vector_type(8)))  float  v8f;

// ---------------- problem sizes ----------------
#define BB 64
#define TT 512
#define DD 512
#define UU 1024
#define KB_U 32   // number of K=32 blocks in the U (recurrent) reduction

// ---------------- workspace layout (bytes) ----------------
#define OFF_WT   33554432
#define OFF_UT   37748736
#define OFF_HB   46137344
#define OFF_BAR  46399488

__device__ __forceinline__ float sigmoidf_(float x) {
  return 1.0f / (1.0f + __expf(-x));
}
__device__ __forceinline__ float tanhf_(float x) {
  x = fminf(fmaxf(x, -15.0f), 15.0f);
  float e = __expf(2.0f * x);
  return (e - 1.0f) / (e + 1.0f);
}

// ---------------------------------------------------------------------------
// Prep: fp32 -> bf16 inputs, transpose weights to bf16, zero h state + barrier
// ---------------------------------------------------------------------------
__global__ void lstm_prep_kernel(const float* __restrict__ inp,
                                 const float* __restrict__ Wf, const float* __restrict__ Wi,
                                 const float* __restrict__ Wc, const float* __restrict__ Wo,
                                 const float* __restrict__ Uf, const float* __restrict__ Ui,
                                 const float* __restrict__ Uc, const float* __restrict__ Uo,
                                 bf16* __restrict__ inp_b, bf16* __restrict__ WT,
                                 bf16* __restrict__ UT, bf16* __restrict__ hbuf,
                                 unsigned int* __restrict__ bar) {
  const size_t N_INP = (size_t)BB * TT * DD;        // 16,777,216
  const size_t N_W   = (size_t)4 * UU * DD;         //  2,097,152
  const size_t N_U   = (size_t)4 * UU * UU;         //  4,194,304
  const size_t N_H   = (size_t)2 * BB * UU;         //    131,072
  const size_t total = N_INP + N_W + N_U + N_H;

  size_t i0 = (size_t)blockIdx.x * blockDim.x + threadIdx.x;
  if (i0 == 0) *bar = 0u;

  const float* Ws[4] = {Wf, Wi, Wc, Wo};
  const float* Us[4] = {Uf, Ui, Uc, Uo};

  size_t stride = (size_t)gridDim.x * blockDim.x;
  for (size_t i = i0; i < total; i += stride) {
    if (i < N_INP) {
      inp_b[i] = (bf16)inp[i];
    } else if (i < N_INP + N_W) {
      size_t j = i - N_INP;
      int    g = (int)(j >> 19);            // / (1024*512)
      size_t r = j & 524287;                // u*512 + d
      size_t u = r >> 9, d = r & 511;
      WT[j] = (bf16)Ws[g][d * UU + u];      // WT[g][u][d] = W[g][d][u]
    } else if (i < N_INP + N_W + N_U) {
      size_t j = i - (N_INP + N_W);
      int    g = (int)(j >> 20);            // / (1024*1024)
      size_t r = j & 1048575;               // n*1024 + k
      size_t n = r >> 10, k = r & 1023;
      UT[j] = (bf16)Us[g][k * UU + n];      // UT[g][n][k] = U[g][k][n]
    } else {
      hbuf[i - (N_INP + N_W + N_U)] = (bf16)0.0f;
    }
  }
}

// ---------------------------------------------------------------------------
// Persistent recurrent kernel, software-pipelined around the step barrier:
//   acc = xacc(t) + h(t-1)@U  ->  gate fusion, write h(t)  ->  SIGNAL
//   -> compute xacc(t+1) = x(t+1)@W (independent work hides barrier latency)
//   -> WAIT -> next step.
// Grid: 256 blocks (one 16x16 (b,u) tile each), 128 threads (4 gate waves).
// U-weight B-fragments (128 KB) live in LDS; cell state c lives in LDS.
// ---------------------------------------------------------------------------
__global__ void __launch_bounds__(128, 1)
lstm_scan_kernel(const bf16* __restrict__ inp_b, const bf16* __restrict__ WT,
                 const bf16* __restrict__ UT, bf16* __restrict__ hbuf,
                 const float* __restrict__ bfp, const float* __restrict__ bip,
                 const float* __restrict__ bcp, const float* __restrict__ bop,
                 const float* __restrict__ cpe, const float* __restrict__ ope,
                 unsigned int* __restrict__ bar, float* __restrict__ out) {
  __shared__ v16bf bcache[4][KB_U][32];   // 128 KB t-invariant U B fragments
  __shared__ float gacc[4][16][16];
  __shared__ float ctile[16][16];

  const int bt   = blockIdx.x >> 6;   // 0..3  batch tile
  const int ut   = blockIdx.x & 63;   // 0..63 unit tile
  const int wave = threadIdx.x >> 5;  // gate id: 0=f 1=i 2=c 3=o
  const int lane = threadIdx.x & 31;
  const int half = lane >> 4;
  const int m    = lane & 15;         // row within tile (A fragment)
  const int n    = lane & 15;         // col within tile (B fragment)

  const bf16* WTg  = WT + (size_t)wave * (UU * DD);
  const bf16* UTg  = UT + (size_t)wave * (UU * UU);
  const bf16* wrow = WTg + (size_t)(ut * 16 + n) * DD + half * 16;
  const bf16* urow = UTg + (size_t)(ut * 16 + n) * UU + half * 16;
  const bf16* xrow_base = inp_b + (size_t)(bt * 16 + m) * TT * DD + half * 16;

  // ---- one-time fill of the LDS B-fragment cache (U part, t-invariant) ----
#pragma unroll 4
  for (int kb = 0; kb < KB_U; ++kb) {
    bcache[wave][kb][lane] = *(const v16bf*)(urow + kb * 32);
  }

  // Per-element bias / peephole scalars (u fixed per thread for whole scan).
  float bfv[2], biv[2], bcv[2], bov[2], cpv[2], opv[2];
#pragma unroll
  for (int q = 0; q < 2; ++q) {
    int e = threadIdx.x + q * 128;
    int u = ut * 16 + (e & 15);
    bfv[q] = bfp[u]; biv[q] = bip[u]; bcv[q] = bcp[u]; bov[q] = bop[u];
    cpv[q] = cpe[u]; opv[q] = ope[u];
  }
#pragma unroll
  for (int q = 0; q < 2; ++q) {
    int e = threadIdx.x + q * 128;
    ctile[e >> 4][e & 15] = 0.0f;     // c0 = 0
  }
  __syncthreads();

  // ---- prologue: xacc = x(0) @ W ----
  v8f xacc = {0.f, 0.f, 0.f, 0.f, 0.f, 0.f, 0.f, 0.f};
  {
    const bf16* xr = xrow_base;       // t = 0
#pragma unroll 4
    for (int k = 0; k < DD; k += 32) {
      v8bf alo = *(const v8bf*)(xr + k);
      v8bf ahi = *(const v8bf*)(xr + k + 8);
      v16bf a = __builtin_shufflevector(alo, ahi, 0,1,2,3,4,5,6,7,8,9,10,11,12,13,14,15);
      v16bf b = *(const v16bf*)(wrow + k);
      xacc = __builtin_amdgcn_wmma_f32_16x16x32_bf16(false, a, false, b,
                                                     (short)0, xacc, false, false);
    }
  }

  for (int t = 0; t < TT; ++t) {
    const bf16* cur = hbuf + (size_t)(t & 1) * (BB * UU);
    bf16*       nxt = hbuf + (size_t)((t + 1) & 1) * (BB * UU);

    // ---- acc = xacc + h(t-1) @ U ; B from LDS cache ----
    v8f acc = xacc;
    const bf16* hr = cur + (size_t)(bt * 16 + m) * UU + half * 16;
#pragma unroll 4
    for (int kb = 0; kb < KB_U; ++kb) {
      v8bf alo = *(const v8bf*)(hr + kb * 32);
      v8bf ahi = *(const v8bf*)(hr + kb * 32 + 8);
      v16bf a = __builtin_shufflevector(alo, ahi, 0,1,2,3,4,5,6,7,8,9,10,11,12,13,14,15);
      v16bf b = bcache[wave][kb][lane];
      acc = __builtin_amdgcn_wmma_f32_16x16x32_bf16(false, a, false, b,
                                                    (short)0, acc, false, false);
    }

    // Spill this gate's 16x16 tile to LDS (C/D layout: M = r + 8*half, N = lane&15).
#pragma unroll
    for (int r = 0; r < 8; ++r) gacc[wave][r + 8 * half][m] = acc[r];
    __syncthreads();

    // ---- elementwise gate fusion: 256 elems / 128 threads ----
#pragma unroll
    for (int q = 0; q < 2; ++q) {
      int e  = threadIdx.x + q * 128;
      int mm = e >> 4, nn = e & 15;
      float cold = ctile[mm][nn];
      float fg = sigmoidf_(gacc[0][mm][nn] + bfv[q] + cpv[q] * cold);
      float ig = sigmoidf_(gacc[1][mm][nn] + biv[q]);
      float gg = tanhf_(gacc[2][mm][nn] + bcv[q]);
      float cn = fg * cold + ig * gg;
      float og = sigmoidf_(gacc[3][mm][nn] + bov[q] + opv[q] * cn);
      float hv = og * tanhf_(cn);
      ctile[mm][nn] = cn;
      out[((size_t)(bt * 16 + mm) * TT + t) * UU + (ut * 16 + nn)] = hv;
      nxt[(size_t)(bt * 16 + mm) * UU + (ut * 16 + nn)] = (bf16)hv;
    }

    // ---- SIGNAL step barrier ----
    __threadfence();
    __syncthreads();
    if (threadIdx.x == 0) atomicAdd(bar, 1u);

    // ---- overlap: xacc = x(t+1) @ W while the barrier settles ----
    xacc = (v8f){0.f, 0.f, 0.f, 0.f, 0.f, 0.f, 0.f, 0.f};
    if (t + 1 < TT) {
      const bf16* xr = xrow_base + (size_t)(t + 1) * DD;
      if (t + 2 < TT) __builtin_prefetch(xrow_base + (size_t)(t + 2) * DD, 0, 0);
#pragma unroll 4
      for (int k = 0; k < DD; k += 32) {
        v8bf alo = *(const v8bf*)(xr + k);
        v8bf ahi = *(const v8bf*)(xr + k + 8);
        v16bf a = __builtin_shufflevector(alo, ahi, 0,1,2,3,4,5,6,7,8,9,10,11,12,13,14,15);
        v16bf b = *(const v16bf*)(wrow + k);
        xacc = __builtin_amdgcn_wmma_f32_16x16x32_bf16(false, a, false, b,
                                                       (short)0, xacc, false, false);
      }
    }

    // ---- WAIT on step barrier ----
    if (threadIdx.x == 0) {
      unsigned target = (unsigned)(t + 1) * (unsigned)gridDim.x;
      while (atomicAdd(bar, 0u) < target) __builtin_amdgcn_s_sleep(2);
      __threadfence();
    }
    __syncthreads();
  }
}

// ---------------------------------------------------------------------------
extern "C" void kernel_launch(void* const* d_in, const int* in_sizes, int n_in,
                              void* d_out, int out_size, void* d_ws, size_t ws_size,
                              hipStream_t stream) {
  const float* inp = (const float*)d_in[0];
  const float* Wf  = (const float*)d_in[1];
  const float* Uf  = (const float*)d_in[2];
  const float* bfp = (const float*)d_in[3];
  const float* Wi  = (const float*)d_in[4];
  const float* Ui  = (const float*)d_in[5];
  const float* bip = (const float*)d_in[6];
  const float* Wc  = (const float*)d_in[7];
  const float* Uc  = (const float*)d_in[8];
  const float* bcp = (const float*)d_in[9];
  const float* Wo  = (const float*)d_in[10];
  const float* Uo  = (const float*)d_in[11];
  const float* bop = (const float*)d_in[12];
  const float* cpe = (const float*)d_in[13];
  const float* ope = (const float*)d_in[14];
  float* out = (float*)d_out;

  char* ws = (char*)d_ws;
  bf16* inp_b = (bf16*)(ws);
  bf16* WT    = (bf16*)(ws + OFF_WT);
  bf16* UT    = (bf16*)(ws + OFF_UT);
  bf16* hbuf  = (bf16*)(ws + OFF_HB);
  unsigned int* bar = (unsigned int*)(ws + OFF_BAR);

  lstm_prep_kernel<<<4096, 256, 0, stream>>>(inp, Wf, Wi, Wc, Wo, Uf, Ui, Uc, Uo,
                                             inp_b, WT, UT, hbuf, bar);
  lstm_scan_kernel<<<256, 128, 0, stream>>>(inp_b, WT, UT, hbuf,
                                            bfp, bip, bcp, bop, cpe, ope,
                                            bar, out);
}
